// Spectrogram_43860206027231
// MI455X (gfx1250) — matrix operand order repaired and checked by reading, not compiled, not run on hardware
//
#include <hip/hip_runtime.h>

// ---------------------------------------------------------------------------
// Spectrogram (windowed DFT, hop 256, win 1024) as an f16 WMMA GEMM.
//   A: 65536 frames x 1024 samples   (staged per-tile in LDS as f16)
//   B: 1024 x 513 cos + 1024 x 513 sin (converted once to f16, [f][k] major)
//   D: power -> 10*log10 -> per-batch max-subtract -> clamp(-80)
// ---------------------------------------------------------------------------

typedef _Float16 f16;
typedef _Float16 v8h  __attribute__((ext_vector_type(8)));
typedef _Float16 v16h __attribute__((ext_vector_type(16)));
typedef float    v8f  __attribute__((ext_vector_type(8)));

#define N_SIG    64          // 32 batches * 2 channels
#define SIG_LEN  262144
#define N_DFT    1024
#define N_HOP    256
#define N_FILTER 513
#define N_FRAME  1024
#define PAD_L    384         // SAME padding, left

#define ROWS        16       // frames per block tile (M tile)
#define ROW_STRIDE  1032     // f16 elems per LDS row: 1024 + 8 pad -> conflict-free b128
#define STAGE       4864     // union of 16 overlapping frames: 15*256 + 1024

// Order-preserving float <-> uint encoding for atomicMax over signed floats.
__device__ __forceinline__ unsigned enc_f32(float f) {
  unsigned u = __float_as_uint(f);
  return (u & 0x80000000u) ? ~u : (u | 0x80000000u);
}
__device__ __forceinline__ float dec_f32(unsigned u) {
  return (u & 0x80000000u) ? __uint_as_float(u & 0x7fffffffu)
                           : __uint_as_float(~u);
}

// Pass 0: transpose+convert DFT kernels (t*513+f, f32) -> (f*1024+t, f16);
// also reset the 32 per-batch max accumulators (graph-replay safe).
__global__ void spect_prep(const float* __restrict__ kr, const float* __restrict__ ki,
                           f16* __restrict__ br, f16* __restrict__ bi,
                           unsigned* __restrict__ mx) {
  const int idx = blockIdx.x * 256 + threadIdx.x;
  if (blockIdx.x == 0 && threadIdx.x < 32) mx[threadIdx.x] = 0u;
  if (idx < N_FILTER * N_DFT) {
    const int f = idx >> 10;       // filter
    const int t = idx & 1023;      // tap
    br[idx] = (f16)kr[t * N_FILTER + f];
    bi[idx] = (f16)ki[t * N_FILTER + f];
  }
}

// Pass 1: WMMA GEMM + power + log10 + per-batch max (atomic).
__global__ __launch_bounds__(256) void spect_main(
    const float* __restrict__ x, const f16* __restrict__ br,
    const f16* __restrict__ bi, float* __restrict__ out,
    unsigned* __restrict__ mx) {
  __shared__ __align__(32) f16 sA[ROWS * ROW_STRIDE];   // 33 KB of 320 KB/WGP

  const int tid  = threadIdx.x;
  const int lane = tid & 31;
  const int wave = tid >> 5;
  const int ng   = blockIdx.x;    // filter group of 128
  const int ft   = blockIdx.y;    // frame tile (16 frames)
  const int sig  = blockIdx.z;    // signal (batch*2 + channel)

  // ---- stage 16 overlapping frames into padded f16 rows ----
  const size_t sb = (size_t)sig * SIG_LEN;
  const int w0 = ft * (ROWS * N_HOP) - PAD_L;           // window start (may be <0)
  for (int i = tid; i < STAGE; i += 256) {
    const int g = w0 + i;
    float v = 0.0f;
    if (g >= 0 && g < SIG_LEN) v = x[sb + g];
    const f16 h = (f16)v;
    int mlo = (i - 768) >> 8; if (mlo < 0)  mlo = 0;    // ceil((i-1023)/256)
    int mhi = i >> 8;         if (mhi > 15) mhi = 15;
    for (int m = mlo; m <= mhi; ++m)
      sA[m * ROW_STRIDE + (i - (m << 8))] = h;
  }
  __syncthreads();

  const int ncol = lane & 15;
  const int hihalf = lane >> 4;                         // 0: lanes 0-15, 1: 16-31
  int f = ng * 128 + wave * 16 + ncol;
  const bool valid = (f < N_FILTER);
  if (f > N_FILTER - 1) f = N_FILTER - 1;               // clamp: safe loads, guarded stores

  // B fragment: 16 contiguous K per lane, K base = lane<16 ? 0 : 16 (32B loads)
  const f16* pR = br + (size_t)f * N_DFT + (hihalf << 4);
  const f16* pI = bi + (size_t)f * N_DFT + (hihalf << 4);
  // A fragment: row = lane%16, k0 = lane<16 ? 0 : 16; halves 0-7 -> K k0..k0+7,
  // halves 8-15 -> K k0+8..k0+15  => two 16B LDS b128 loads.
  const f16* pA = sA + (lane & 15) * ROW_STRIDE + (hihalf << 4);

  v8f accR = {};
  v8f accI = {};
#pragma unroll 4
  for (int kk = 0; kk < N_DFT / 32; ++kk) {
    const v8h alo = *(const v8h*)(pA + kk * 32);
    const v8h ahi = *(const v8h*)(pA + kk * 32 + 8);
    const v16h a = __builtin_shufflevector(
        alo, ahi, 0, 1, 2, 3, 4, 5, 6, 7, 8, 9, 10, 11, 12, 13, 14, 15);
    const v16h vbr = *(const v16h*)(pR + kk * 32);
    const v16h vbi = *(const v16h*)(pI + kk * 32);
    accR = __builtin_amdgcn_wmma_f32_16x16x32_f16(false, a, false, vbr,
                                                  (short)0, accR, false, false);
    accI = __builtin_amdgcn_wmma_f32_16x16x32_f16(false, a, false, vbi,
                                                  (short)0, accI, false, false);
  }

  // ---- fused epilogue: power -> dB, store, per-batch max ----
  float lmax = -3.0e38f;
  if (valid) {
    const int b   = sig >> 1;
    const int ch  = sig & 1;
    const int fr0 = (ft << 4) + (hihalf << 3);          // D: VGPR v -> frame fr0+v
    size_t base = (((size_t)b * N_FILTER + (size_t)f) * N_FRAME + fr0) * 2 + ch;
#pragma unroll
    for (int v = 0; v < 8; ++v) {
      float p = accR[v] * accR[v] + accI[v] * accI[v];
      p = fmaxf(p, 1e-10f);
      const float lg = 10.0f * __log10f(p);
      out[base + (size_t)(v * 2)] = lg;
      lmax = fmaxf(lmax, lg);
    }
  }
#pragma unroll
  for (int off = 16; off > 0; off >>= 1)
    lmax = fmaxf(lmax, __shfl_xor(lmax, off, 32));
  if (lane == 0) atomicMax(&mx[sig >> 1], enc_f32(lmax));
}

// Pass 2: subtract per-batch max, clamp at -80 dB.
__global__ void spect_final(float* __restrict__ out,
                            const unsigned* __restrict__ mx) {
  const size_t total = (size_t)32 * N_FILTER * N_FRAME * 2;
  const size_t idx = (size_t)blockIdx.x * 256 + threadIdx.x;
  if (idx >= total) return;
  const unsigned per_b = (unsigned)(N_FILTER * N_FRAME * 2);
  const int b = (int)(idx / per_b);
  const float m = dec_f32(mx[b]);
  out[idx] = fmaxf(out[idx] - m, -80.0f);
}

extern "C" void kernel_launch(void* const* d_in, const int* in_sizes, int n_in,
                              void* d_out, int out_size, void* d_ws, size_t ws_size,
                              hipStream_t stream) {
  (void)in_sizes; (void)n_in; (void)out_size; (void)ws_size;
  const float* x  = (const float*)d_in[0];
  const float* kr = (const float*)d_in[1];
  const float* ki = (const float*)d_in[2];
  float* out = (float*)d_out;

  // Workspace: [0,128) per-batch max slots; f16 kernels at +4096 (aligned).
  unsigned* mx = (unsigned*)d_ws;
  f16* br = (f16*)((char*)d_ws + 4096);
  f16* bi = br + (size_t)N_FILTER * N_DFT;

  spect_prep<<<(N_FILTER * N_DFT + 255) / 256, 256, 0, stream>>>(kr, ki, br, bi, mx);

  dim3 grid(5 /* ceil(513/128) */, N_FRAME / ROWS, N_SIG);
  spect_main<<<grid, 256, 0, stream>>>(x, br, bi, out, mx);

  const long long total = (long long)32 * N_FILTER * N_FRAME * 2;
  spect_final<<<(unsigned)((total + 255) / 256), 256, 0, stream>>>(out, mx);
}
